// Transliterator_7687991460205
// MI455X (gfx1250) — compile-verified, matrix operand order checked by reading, not run on hardware
//
#include <hip/hip_runtime.h>
#include <hip/hip_bf16.h>
#include <stdint.h>

typedef __bf16 bf16;
typedef __attribute__((ext_vector_type(16))) __bf16 v16bf;
typedef __attribute__((ext_vector_type(8)))  __bf16 v8bf;
typedef __attribute__((ext_vector_type(8)))  float  v8f;

#define S_LEN  32
#define T_LEN  48
#define BATCH  1024
#define E_DIM  256
#define H_DIM  1024
#define VT_DIM 256
#define G4H    4096   // 4*H

// ---------------------------------------------------------------------------
// CDNA5 primitives
// ---------------------------------------------------------------------------
static __device__ __forceinline__ v8f wmma_bf16(v16bf a, v16bf b, v8f c) {
  return __builtin_amdgcn_wmma_f32_16x16x32_bf16(false, a, false, b, (short)0, c,
                                                 false, false);
}

// Async global->LDS copy, 16B per lane (ASYNCcnt-tracked; no VGPR staging).
static __device__ __forceinline__ void async_cp16(unsigned lds_off, const bf16* g) {
  asm volatile("global_load_async_to_lds_b128 %0, %1, off"
               :: "v"(lds_off), "v"(g) : "memory");
}
static __device__ __forceinline__ void wait_async0() {
  asm volatile("s_wait_asynccnt 0x0" ::: "memory");
}

static __device__ __forceinline__ float sigmoidf_(float x) {
  return 1.0f / (1.0f + __expf(-x));
}
// Branchless tanh: 2*sigmoid(2x) - 1 (avoids libm's divergent exec-mask path)
static __device__ __forceinline__ float tanh_fast(float x) {
  return __builtin_fmaf(2.0f, sigmoidf_(2.0f * x), -1.0f);
}

// A fragment (16x32 bf16) read out of an LDS tile stored row-major [rows][32 k].
static __device__ __forceinline__ v16bf lds_a_frag(const bf16* tile, int row0, int lane) {
  const bf16* p = tile + (size_t)(row0 + (lane & 15)) * 32 + ((lane >> 4) << 3);
  union { v16bf v; v8bf h[2]; } u;
  u.h[0] = *(const v8bf*)(p);
  u.h[1] = *(const v8bf*)(p + 16);
  return u.v;
}

// ---------------------------------------------------------------------------
// Weight pack: fp32 -> bf16 in B-fragment-native order.
// w_is_nk==1 : W is [N,K] row-major (z = x @ W^T; B(k,n) = W[n*K+k])   (LSTM W)
// w_is_nk==0 : W is [K,N] row-major (z = x @ W;   B(k,n) = W[k*N+n])   (fc_W)
// Packed: [(nTile*KT + kTile)][lane 0..31][16 bf16 contiguous]
// ---------------------------------------------------------------------------
__global__ __launch_bounds__(256) void pack_b_kernel(const float* __restrict__ W,
                                                     bf16* __restrict__ out,
                                                     int N, int K, int w_is_nk) {
  const int tid   = blockIdx.x * blockDim.x + threadIdx.x;
  const int total = N * K;
  if (tid >= total) return;
  const int e     = tid & 7;
  const int chunk = (tid >> 3) & 1;
  const int lane  = (tid >> 4) & 31;
  const int grp   = tid >> 9;          // nTile*KT + kTile
  const int KT    = K >> 5;
  const int kTile = grp % KT;
  const int nTile = grp / KT;
  const int n = (nTile << 4) + (lane & 15);
  const int k = (kTile << 5) + ((lane >> 4) << 3) + (chunk << 4) + e;
  const float v = w_is_nk ? W[(size_t)n * K + k] : W[(size_t)k * N + n];
  out[tid] = (bf16)v;
}

// Embedding gather + fp32->bf16 convert: out[r, :] = emb[tok[r], :]
__global__ __launch_bounds__(256) void gather_emb_kernel(const int* __restrict__ tok,
                                                         const float* __restrict__ emb,
                                                         bf16* __restrict__ out,
                                                         int nrows) {
  const int tid = blockIdx.x * blockDim.x + threadIdx.x;
  if (tid >= nrows * E_DIM) return;
  const int e = tid & (E_DIM - 1);
  const int r = tid >> 8;              // E_DIM == 256
  out[tid] = (bf16)emb[(size_t)tok[r] * E_DIM + e];
}

__global__ __launch_bounds__(256) void zero_f32_kernel(float* p, int n) {
  const int t = blockIdx.x * blockDim.x + threadIdx.x;
  if (t < n) p[t] = 0.0f;
}
__global__ __launch_bounds__(256) void zero_bf16_kernel(bf16* p, int n) {
  const int t = blockIdx.x * blockDim.x + threadIdx.x;
  if (t < n) p[t] = (bf16)0.0f;
}

// ---------------------------------------------------------------------------
// Fused LSTM step: z = x @ Wih^T + h @ Whh^T + b ; gate math ; update h,c.
// Grid (BATCH/64, H/64), 256 threads = 8 wave32 arranged 4(M) x 2(N).
// Per wave: 16 rows x 32 cols x 4 gates = 8 WMMA accumulators; 8 independent
// back-to-back WMMAs per K slice sharing one A fragment.
// Double-buffered LDS pipeline fed by global_load_async_to_lds_b128;
// x-GEMM and h-GEMM fused into one K pipeline (KTx + KTh slices).
// ---------------------------------------------------------------------------
__global__ __launch_bounds__(256) void lstm_step_kernel(
    const bf16*  __restrict__ x,     int Kx,          // [BATCH, Kx] bf16
    const bf16*  __restrict__ h_in,                   // [BATCH, H]  bf16
    float*       __restrict__ c,                      // [BATCH, H]  f32 (in/out)
    bf16*        __restrict__ h_out,                  // [BATCH, H]  bf16
    const bf16*  __restrict__ Wih_p,                  // packed, N=4H, K=Kx
    const bf16*  __restrict__ Whh_p,                  // packed, N=4H, K=H
    const float* __restrict__ bias)                   // [4H]
{
  __shared__ bf16 sA[2][64 * 32];       // 2 x  4KB : A tile, row-major [64][32]
  __shared__ bf16 sB[2][16 * 512];      // 2 x 16KB : 16 B-frag tiles (4 gates x 4 nt)

  const int tid  = threadIdx.x;
  const int lane = tid & 31;
  const int wave = tid >> 5;
  const int wm   = wave & 3;                          // 0..3 (M)
  const int wn   = wave >> 2;                         // 0..1 (N half)
  const int mBlk = blockIdx.x * 64;
  const int ntBase = blockIdx.y * 4;                  // B tile base within a gate
  const int NTG  = H_DIM >> 4;                        // 64 tiles per gate

  const int KTx = Kx >> 5;
  const int KTh = H_DIM >> 5;
  const int KT  = KTx + KTh;

  const unsigned sA0 = (unsigned)(uintptr_t)(&sA[0][0]);
  const unsigned sB0 = (unsigned)(uintptr_t)(&sB[0][0]);

  // per-thread copy assignment
  const int aRow = tid >> 2;                          // 0..63
  const int aCh  = tid & 3;                           // 0..3 (16B chunk in row)

  auto issue = [&](int kt, int buf) {
    const bf16* asrc; int ald; const bf16* wsrc; int ktl; int KTl;
    if (kt < KTx) { asrc = x;    ald = Kx;    wsrc = Wih_p; ktl = kt;       KTl = KTx; }
    else          { asrc = h_in; ald = H_DIM; wsrc = Whh_p; ktl = kt - KTx; KTl = KTh; }
    // A tile: 64 rows x 32 k -> 256 x 16B chunks, 1 per thread
    {
      const bf16* g = asrc + (size_t)(mBlk + aRow) * ald + (ktl << 5) + (aCh << 3);
      const unsigned l = sA0 + (unsigned)(buf * (64 * 32 * 2)) +
                         (unsigned)((aRow << 6) + (aCh << 4));
      async_cp16(l, g);
    }
    // B tiles: 16 x 1KB contiguous -> 1024 x 16B chunks, 4 per thread
#pragma unroll
    for (int i = 0; i < 4; ++i) {
      const int ch  = tid + (i << 8);                 // 0..1023
      const int tI  = ch >> 6;                        // tile 0..15 = gate*4 + nt
      const int o16 = ch & 63;                        // 16B chunk within tile
      const size_t nTile = (size_t)(tI >> 2) * NTG + (size_t)(ntBase + (tI & 3));
      const bf16* g = wsrc + (nTile * KTl + ktl) * 512 + (o16 << 3);
      const unsigned l = sB0 + (unsigned)(buf * (16 * 512 * 2)) +
                         (unsigned)((tI << 10) + (o16 << 4));
      async_cp16(l, g);
    }
  };

  v8f acc[4][2] = {};                                 // [gate][n-subtile]

  issue(0, 0);
  wait_async0();
  __syncthreads();

  for (int kt = 0; kt < KT; ++kt) {
    const int cur = kt & 1;
    if (kt + 1 < KT) issue(kt + 1, cur ^ 1);          // prefetch next slice

    const v16bf a = lds_a_frag(&sA[cur][0], wm << 4, lane);
    const bf16* bb = &sB[cur][lane << 4];
#pragma unroll
    for (int g = 0; g < 4; ++g) {
#pragma unroll
      for (int j = 0; j < 2; ++j) {
        const int tI = (g << 2) + (wn << 1) + j;
        acc[g][j] = wmma_bf16(a, *(const v16bf*)(bb + (tI << 9)), acc[g][j]);
      }
    }

    wait_async0();                                    // next slice landed
    __syncthreads();                                  // everyone done read+write
  }

  // Epilogue: C/D layout — VGPR r holds (row m0 + r + 8*(lane>>4), col n0 + lane&15).
  const int m0 = mBlk + (wm << 4);
#pragma unroll
  for (int j = 0; j < 2; ++j) {
    const int   col = blockIdx.y * 64 + ((wn << 1) + j) * 16 + (lane & 15);
    const float bi  = bias[0 * H_DIM + col];
    const float bf_ = bias[1 * H_DIM + col];
    const float bg  = bias[2 * H_DIM + col];
    const float bo  = bias[3 * H_DIM + col];
#pragma unroll
    for (int r = 0; r < 8; ++r) {
      const int    row = m0 + r + ((lane >> 4) << 3);
      const size_t idx = (size_t)row * H_DIM + col;
      const float ig = sigmoidf_(acc[0][j][r] + bi);
      const float fg = sigmoidf_(acc[1][j][r] + bf_);
      const float gg = tanh_fast(acc[2][j][r] + bg);
      const float og = sigmoidf_(acc[3][j][r] + bo);
      const float cn = fg * c[idx] + ig * gg;
      c[idx]     = cn;
      h_out[idx] = (bf16)(og * tanh_fast(cn));
    }
  }
}

// ---------------------------------------------------------------------------
// FC: logits = h @ fc_W + fc_b, [BATCH, VT]. Grid (BATCH/64, VT/32).
// Small (47 launches, K=1024, fully L2-resident) — direct-load WMMA.
// ---------------------------------------------------------------------------
__global__ __launch_bounds__(256) void fc_step_kernel(
    const bf16*  __restrict__ h,      // [BATCH, H] bf16
    const bf16*  __restrict__ Wp,     // packed, N=VT, K=H
    const float* __restrict__ bias,   // [VT]
    float*       __restrict__ out)    // [BATCH, VT]
{
  const int lane = threadIdx.x & 31;
  const int wave = threadIdx.x >> 5;
  const int wm   = wave & 3;
  const int wn   = wave >> 2;
  const int m0   = blockIdx.x * 64 + wm * 16;
  const int n0   = blockIdx.y * 32 + wn * 16;
  const int nt   = n0 >> 4;
  const int KT   = H_DIM >> 5;

  v8f acc = {};
  for (int kt = 0; kt < KT; ++kt) {
    const bf16* ap = h + (size_t)(m0 + (lane & 15)) * H_DIM + (kt << 5) + ((lane >> 4) << 3);
    union { v16bf v; v8bf hh[2]; } ua;
    ua.hh[0] = *(const v8bf*)(ap);
    ua.hh[1] = *(const v8bf*)(ap + 16);
    const v16bf b = *(const v16bf*)(Wp + ((((size_t)nt * KT + kt) * 32 + lane) << 4));
    acc = wmma_bf16(ua.v, b, acc);
  }
  const int   col = n0 + (lane & 15);
  const float bb  = bias[col];
#pragma unroll
  for (int r = 0; r < 8; ++r) {
    const int row = m0 + r + ((lane >> 4) << 3);
    out[(size_t)row * VT_DIM + col] = acc[r] + bb;
  }
}

// ---------------------------------------------------------------------------
// Host orchestration
// ---------------------------------------------------------------------------
extern "C" void kernel_launch(void* const* d_in, const int* in_sizes, int n_in,
                              void* d_out, int out_size, void* d_ws, size_t ws_size,
                              hipStream_t stream) {
  (void)in_sizes; (void)n_in; (void)out_size; (void)ws_size;

  const int*   src      = (const int*)  d_in[0];
  const int*   tgt      = (const int*)  d_in[1];
  const float* enc_emb  = (const float*)d_in[2];
  const float* enc_Wih0 = (const float*)d_in[3];
  const float* enc_Whh0 = (const float*)d_in[4];
  const float* enc_b0   = (const float*)d_in[5];
  const float* enc_Wih1 = (const float*)d_in[6];
  const float* enc_Whh1 = (const float*)d_in[7];
  const float* enc_b1   = (const float*)d_in[8];
  const float* dec_emb  = (const float*)d_in[9];
  const float* dec_Wih0 = (const float*)d_in[10];
  const float* dec_Whh0 = (const float*)d_in[11];
  const float* dec_b0   = (const float*)d_in[12];
  const float* dec_Wih1 = (const float*)d_in[13];
  const float* dec_Whh1 = (const float*)d_in[14];
  const float* dec_b1   = (const float*)d_in[15];
  const float* fc_W     = (const float*)d_in[16];
  const float* fc_b     = (const float*)d_in[17];
  float* out = (float*)d_out;

  // ---- carve workspace ----
  char* ws = (char*)d_ws;
  size_t off = 0;
  auto carve = [&](size_t bytes) -> char* {
    char* p = ws + off;
    off = (off + bytes + 255) & ~(size_t)255;
    return p;
  };
  bf16* pWih_e0 = (bf16*)carve((size_t)G4H * E_DIM * sizeof(bf16));
  bf16* pWhh_e0 = (bf16*)carve((size_t)G4H * H_DIM * sizeof(bf16));
  bf16* pWih_e1 = (bf16*)carve((size_t)G4H * H_DIM * sizeof(bf16));
  bf16* pWhh_e1 = (bf16*)carve((size_t)G4H * H_DIM * sizeof(bf16));
  bf16* pWih_d0 = (bf16*)carve((size_t)G4H * E_DIM * sizeof(bf16));
  bf16* pWhh_d0 = (bf16*)carve((size_t)G4H * H_DIM * sizeof(bf16));
  bf16* pWih_d1 = (bf16*)carve((size_t)G4H * H_DIM * sizeof(bf16));
  bf16* pWhh_d1 = (bf16*)carve((size_t)G4H * H_DIM * sizeof(bf16));
  bf16* pFc     = (bf16*)carve((size_t)H_DIM * VT_DIM * sizeof(bf16));
  bf16* xe      = (bf16*)carve((size_t)S_LEN * BATCH * E_DIM * sizeof(bf16));
  bf16* xd      = (bf16*)carve((size_t)(T_LEN - 1) * BATCH * E_DIM * sizeof(bf16));
  bf16* h0buf[2]; float* c0;
  bf16* h1buf[2]; float* c1;
  h0buf[0] = (bf16*)carve((size_t)BATCH * H_DIM * sizeof(bf16));
  h0buf[1] = (bf16*)carve((size_t)BATCH * H_DIM * sizeof(bf16));
  c0       = (float*)carve((size_t)BATCH * H_DIM * sizeof(float));
  h1buf[0] = (bf16*)carve((size_t)BATCH * H_DIM * sizeof(bf16));
  h1buf[1] = (bf16*)carve((size_t)BATCH * H_DIM * sizeof(bf16));
  c1       = (float*)carve((size_t)BATCH * H_DIM * sizeof(float));

  auto pack = [&](const float* W, bf16* dst, int N, int K, int w_is_nk) {
    const int total = N * K;
    pack_b_kernel<<<(total + 255) / 256, 256, 0, stream>>>(W, dst, N, K, w_is_nk);
  };

  // ---- 1) pack weights to bf16 WMMA-B layout (L2-resident thereafter) ----
  pack(enc_Wih0, pWih_e0, G4H, E_DIM, 1);
  pack(enc_Whh0, pWhh_e0, G4H, H_DIM, 1);
  pack(enc_Wih1, pWih_e1, G4H, H_DIM, 1);
  pack(enc_Whh1, pWhh_e1, G4H, H_DIM, 1);
  pack(dec_Wih0, pWih_d0, G4H, E_DIM, 1);
  pack(dec_Whh0, pWhh_d0, G4H, H_DIM, 1);
  pack(dec_Wih1, pWih_d1, G4H, H_DIM, 1);
  pack(dec_Whh1, pWhh_d1, G4H, H_DIM, 1);
  pack(fc_W,     pFc,     VT_DIM, H_DIM, 0);   // fc_W is [K=H, N=VT]

  // ---- 2) embedding gathers (fp32 -> bf16) ----
  {
    const int rows_e = S_LEN * BATCH;
    gather_emb_kernel<<<(rows_e * E_DIM + 255) / 256, 256, 0, stream>>>(
        src, enc_emb, xe, rows_e);
    const int rows_d = (T_LEN - 1) * BATCH;    // tgt[:-1]
    gather_emb_kernel<<<(rows_d * E_DIM + 255) / 256, 256, 0, stream>>>(
        tgt, dec_emb, xd, rows_d);
  }

  // ---- 3) zero initial states and out[0] row ----
  {
    const int nh = BATCH * H_DIM;
    zero_bf16_kernel<<<(nh + 255) / 256, 256, 0, stream>>>(h0buf[0], nh);
    zero_bf16_kernel<<<(nh + 255) / 256, 256, 0, stream>>>(h1buf[0], nh);
    zero_f32_kernel<<<(nh + 255) / 256, 256, 0, stream>>>(c0, nh);
    zero_f32_kernel<<<(nh + 255) / 256, 256, 0, stream>>>(c1, nh);
    const int n0 = BATCH * VT_DIM;
    zero_f32_kernel<<<(n0 + 255) / 256, 256, 0, stream>>>(out, n0);
  }

  const dim3 lstm_grid(BATCH / 64, H_DIM / 64);
  const dim3 fc_grid(BATCH / 64, VT_DIM / 32);
  int p0 = 0, p1 = 0;   // ping-pong parity for layer-0 / layer-1 h buffers

  // ---- 4) encoder: 2-layer LSTM, stepwise interleaved ----
  for (int t = 0; t < S_LEN; ++t) {
    lstm_step_kernel<<<lstm_grid, 256, 0, stream>>>(
        xe + (size_t)t * BATCH * E_DIM, E_DIM,
        h0buf[p0], c0, h0buf[1 - p0], pWih_e0, pWhh_e0, enc_b0);
    p0 ^= 1;
    lstm_step_kernel<<<lstm_grid, 256, 0, stream>>>(
        h0buf[p0], H_DIM,
        h1buf[p1], c1, h1buf[1 - p1], pWih_e1, pWhh_e1, enc_b1);
    p1 ^= 1;
  }

  // ---- 5) decoder: states continue from encoder finals (same buffers) ----
  for (int t = 0; t < T_LEN - 1; ++t) {
    lstm_step_kernel<<<lstm_grid, 256, 0, stream>>>(
        xd + (size_t)t * BATCH * E_DIM, E_DIM,
        h0buf[p0], c0, h0buf[1 - p0], pWih_d0, pWhh_d0, dec_b0);
    p0 ^= 1;
    lstm_step_kernel<<<lstm_grid, 256, 0, stream>>>(
        h0buf[p0], H_DIM,
        h1buf[p1], c1, h1buf[1 - p1], pWih_d1, pWhh_d1, dec_b1);
    p1 ^= 1;
    fc_step_kernel<<<fc_grid, 256, 0, stream>>>(
        h1buf[p1], pFc, fc_b, out + (size_t)(t + 1) * BATCH * VT_DIM);
  }
}
